// SetAbstractionLayer_21638045237344
// MI455X (gfx1250) — matrix operand order repaired and checked by reading, not compiled, not run on hardware
//
#include <hip/hip_runtime.h>
#include <hip/hip_bf16.h>
#include <cstdint>

// ---------------------------------------------------------------------------
// Problem constants (from reference)
// ---------------------------------------------------------------------------
#define BATCH 16
#define NPTS  4096
#define DIM   64
#define HID   256
#define NSAMP 1024
#define KNN_K 8

#define OUT_ELEMS  (BATCH * NPTS * DIM)        // 4194304 floats
#define KNN_ELEMS  (BATCH * NSAMP * KNN_K)     // 131072 ints
#define FPS_ELEMS  (BATCH * NSAMP)             // 16384 ints

typedef __attribute__((ext_vector_type(16))) _Float16 v16h;
typedef __attribute__((ext_vector_type(8)))  float    v8f;
typedef __attribute__((ext_vector_type(4)))  unsigned int v4u;
typedef __attribute__((ext_vector_type(8)))  int      v8i;
typedef __attribute__((ext_vector_type(4)))  int      v4i;

// ---------------------------------------------------------------------------
// WMMA helpers (CDNA5 V_WMMA_F32_16X16X32_F16, wave32)
// ---------------------------------------------------------------------------
__device__ inline v8f wmma_f16(v16h a, v16h b, v8f c) {
  // 8 args: (neg_a, A, neg_b, B, c_mod, C, reuse_a, reuse_b)
  return __builtin_amdgcn_wmma_f32_16x16x32_f16(false, a, false, b,
                                                (short)0, c, false, false);
}

// A-matrix fragment: 16x32 f16, row-major source with leading dim `ld`.
// Layout: lane L<16 holds row L, K in {0..15};
// lane L+16 holds row L, K in {16..31}; 2 halves per VGPR.
__device__ inline v16h load_a_f16(const _Float16* __restrict__ src, int ld,
                                  int row0, int k0, int lane) {
  int r = lane & 15, hi = lane >> 4;
  const _Float16* base = src + (size_t)(row0 + r) * ld + k0;
  v16h a;
#pragma unroll
  for (int v = 0; v < 8; ++v) {
    int k = hi * 16 + 2 * v;
    a[2 * v]     = base[k];
    a[2 * v + 1] = base[k + 1];
  }
  return a;
}

// B-matrix fragment (32x16) from N-major storage (rows = columns n, cols = K),
// i.e. B[k][n] = src[n*ld + k]. lanes 0-15: K=0..15; lanes 16-31: K=16..31.
__device__ inline v16h load_b_nmajor(const _Float16* __restrict__ src, int ld,
                                     int n0, int k0, int lane) {
  int n = lane & 15, hi = lane >> 4;
  const _Float16* base = src + (size_t)(n0 + n) * ld + k0 + hi * 16;
  v16h b;
#pragma unroll
  for (int v = 0; v < 8; ++v) {
    b[2 * v]     = base[2 * v];
    b[2 * v + 1] = base[2 * v + 1];
  }
  return b;
}

// B-matrix fragment (32x16) from K-major storage (weights W[k][n], ld = N).
__device__ inline v16h load_b_kmajor(const _Float16* __restrict__ w, int ldn,
                                     int n0, int k0, int lane) {
  int n = lane & 15, hi = lane >> 4;
  const _Float16* base = w + (size_t)(k0 + hi * 16) * ldn + (n0 + n);
  v16h b;
#pragma unroll
  for (int v = 0; v < 8; ++v) {
    b[2 * v]     = base[(size_t)(2 * v) * ldn];
    b[2 * v + 1] = base[(size_t)(2 * v + 1) * ldn];
  }
  return b;
}

// ---------------------------------------------------------------------------
// Tensor Data Mover: DMA a 2-D f16 tile (rows x rowlen, row stride in elems)
// from global memory into LDS. Issued by one wave; completion via TENSORcnt.
// D# layout per CDNA5 ISA ch.8 (group0: count/lds/global/type, group1: dims).
// ---------------------------------------------------------------------------
__device__ inline void tdm_load_2d_f16(unsigned lds_off, const void* gptr,
                                       unsigned rows, unsigned rowlen,
                                       unsigned stride0) {
  unsigned long long ga = (unsigned long long)(uintptr_t)gptr;
  v4u g0;
  g0[0] = 1u;                                      // count=1, user descriptor
  g0[1] = lds_off;                                 // lds_addr (bytes)
  g0[2] = (unsigned)(ga & 0xFFFFFFFFu);            // global_addr[31:0]
  g0[3] = (unsigned)((ga >> 32) & 0x01FFFFFFu)     // global_addr[56:32]
        | (2u << 30);                              // type = 2 ("image")
  v8i g1;
  g1[0] = (int)(1u << 16);                         // data_size = 2 bytes
  g1[1] = (int)((rowlen & 0xFFFFu) << 16);         // tensor_dim0[15:0]
  g1[2] = (int)((rowlen >> 16) | ((rows & 0xFFFFu) << 16));   // dim0 hi, dim1 lo
  g1[3] = (int)((rows >> 16) | ((rowlen & 0xFFFFu) << 16));   // dim1 hi, tile_dim0
  g1[4] = (int)(rows & 0xFFFFu);                   // tile_dim1 (tile_dim2 = 0)
  g1[5] = (int)stride0;                            // tensor_dim0_stride[31:0]
  g1[6] = 0;                                       // stride0 hi, stride1 lo
  g1[7] = 0;                                       // stride1 hi
  v4i gz = {0, 0, 0, 0};                           // groups 2/3 unused (2-D)
#if defined(__clang_major__) && (__clang_major__ >= 23)
  v8i gz8 = {0, 0, 0, 0, 0, 0, 0, 0};
  __builtin_amdgcn_tensor_load_to_lds(g0, g1, gz, gz, gz8, 0);
#else
  __builtin_amdgcn_tensor_load_to_lds(g0, g1, gz, gz, 0);
#endif
}

// ---------------------------------------------------------------------------
// Small utilities
// ---------------------------------------------------------------------------
__device__ inline unsigned long long shfl_xor_u64(unsigned long long v, int mask) {
  unsigned lo = (unsigned)v;
  unsigned hi = (unsigned)(v >> 32);
  lo = __shfl_xor(lo, mask, 32);
  hi = __shfl_xor(hi, mask, 32);
  return ((unsigned long long)hi << 32) | lo;
}

// Maintain ascending top-8 (smallest distances first).
__device__ inline void topk_insert(float d, int idx, float (&bd)[8], int (&bi)[8]) {
  if (d < bd[7]) {
    bd[7] = d; bi[7] = idx;
#pragma unroll
    for (int e = 7; e > 0; --e) {
      if (bd[e] < bd[e - 1]) {
        float td = bd[e]; bd[e] = bd[e - 1]; bd[e - 1] = td;
        int   ti = bi[e]; bi[e] = bi[e - 1]; bi[e - 1] = ti;
      }
    }
  }
}

// ---------------------------------------------------------------------------
// Kernel 1: convert points to f16 + squared norms
// ---------------------------------------------------------------------------
__global__ __launch_bounds__(256) void prep_points(const float* __restrict__ pts,
                                                   _Float16* __restrict__ pts_h,
                                                   float* __restrict__ norms) {
  int p = blockIdx.x * blockDim.x + threadIdx.x;  // 0 .. B*N-1
  const float4* src = (const float4*)(pts + (size_t)p * DIM);
  float acc = 0.f;
#pragma unroll
  for (int q = 0; q < DIM / 4; ++q) {
    float4 v = src[q];
    acc += v.x * v.x + v.y * v.y + v.z * v.z + v.w * v.w;
    _Float16* dst = pts_h + (size_t)p * DIM + q * 4;
    dst[0] = (_Float16)v.x; dst[1] = (_Float16)v.y;
    dst[2] = (_Float16)v.z; dst[3] = (_Float16)v.w;
  }
  norms[p] = acc;
}

// ---------------------------------------------------------------------------
// Kernel 2: convert weights to f16 (W1: 16384, W2: 65536, W3: 16384 elems)
// ---------------------------------------------------------------------------
__global__ __launch_bounds__(256) void prep_weights(const float* __restrict__ W1,
                                                    const float* __restrict__ W2,
                                                    const float* __restrict__ W3,
                                                    _Float16* __restrict__ w1h,
                                                    _Float16* __restrict__ w2h,
                                                    _Float16* __restrict__ w3h) {
  int i = blockIdx.x * blockDim.x + threadIdx.x;  // < 98304
  if (i < 16384)       w1h[i]         = (_Float16)W1[i];
  else if (i < 81920)  w2h[i - 16384] = (_Float16)W2[i - 16384];
  else if (i < 98304)  w3h[i - 81920] = (_Float16)W3[i - 81920];
}

// ---------------------------------------------------------------------------
// Kernel 3: farthest point sampling. One workgroup (1024 thr / 32 waves) per
// batch; per-thread running min over 4 points; block argmax via packed u64.
// ---------------------------------------------------------------------------
__global__ __launch_bounds__(1024) void fps_kernel(const float* __restrict__ pts,
                                                   int* __restrict__ fps_ws,
                                                   int* __restrict__ fps_out) {
  const int b = blockIdx.x;
  const int t = threadIdx.x;
  const int lane = t & 31, wave = t >> 5;
  __shared__ float4 cent4[DIM / 4];
  __shared__ unsigned long long wred[32];
  __shared__ int cur;

  const float4* p4 = (const float4*)(pts + (size_t)b * NPTS * DIM);
  float mind[4] = {3.402823466e38f, 3.402823466e38f, 3.402823466e38f, 3.402823466e38f};

  if (t == 0) {
    cur = 0;
    fps_ws[(size_t)b * NSAMP] = 0;
    fps_out[(size_t)b * NSAMP] = 0;
  }
  __syncthreads();

  for (int s = 1; s < NSAMP; ++s) {
    if (t < DIM / 4) cent4[t] = p4[(size_t)cur * (DIM / 4) + t];
    __syncthreads();

    unsigned long long best = 0ull;
#pragma unroll
    for (int i = 0; i < 4; ++i) {
      int p = t + i * 1024;
      float acc = 0.f;
#pragma unroll
      for (int q = 0; q < DIM / 4; ++q) {
        float4 v = p4[(size_t)p * (DIM / 4) + q];
        float4 c = cent4[q];
        float dx = v.x - c.x, dy = v.y - c.y, dz = v.z - c.z, dw = v.w - c.w;
        acc += dx * dx + dy * dy + dz * dz + dw * dw;
      }
      mind[i] = fminf(mind[i], acc);
      unsigned long long key =
          ((unsigned long long)__float_as_uint(mind[i]) << 32) |
          (unsigned)(0xFFFFFFFFu - (unsigned)p);  // prefer smaller idx on ties
      best = (key > best) ? key : best;
    }
#pragma unroll
    for (int m = 16; m >= 1; m >>= 1) {
      unsigned long long o = shfl_xor_u64(best, m);
      best = (o > best) ? o : best;
    }
    if (lane == 0) wred[wave] = best;
    __syncthreads();
    if (wave == 0) {
      unsigned long long k = wred[lane];
#pragma unroll
      for (int m = 16; m >= 1; m >>= 1) {
        unsigned long long o = shfl_xor_u64(k, m);
        k = (o > k) ? o : k;
      }
      if (lane == 0) {
        int idx = (int)(0xFFFFFFFFu - (unsigned)(k & 0xFFFFFFFFull));
        cur = idx;
        fps_ws[(size_t)b * NSAMP + s] = idx;
        fps_out[(size_t)b * NSAMP + s] = idx;
      }
    }
    __syncthreads();
  }
}

// ---------------------------------------------------------------------------
// Kernel 4: gather centroid f16 rows
// ---------------------------------------------------------------------------
__global__ __launch_bounds__(256) void gather_cent(const int* __restrict__ fps_ws,
                                                   const _Float16* __restrict__ pts_h,
                                                   _Float16* __restrict__ cent_h) {
  int sg = blockIdx.x * blockDim.x + threadIdx.x;  // < B*S
  int b = sg >> 10;
  int idx = fps_ws[sg];
  const float4* src = (const float4*)(pts_h + ((size_t)b * NPTS + idx) * DIM);
  float4* dst = (float4*)(cent_h + (size_t)sg * DIM);
#pragma unroll
  for (int q = 0; q < 8; ++q) dst[q] = src[q];
}

// ---------------------------------------------------------------------------
// Kernel 5: KNN. Block = 128 (4 waves) handles 16 centroids of one batch.
// Centroid tile staged LDS-side by the Tensor Data Mover; d = |c|^2 - 2 c.p
// + |p|^2 with c.p from WMMA (K=64 -> 2x 16x16x32); fused streaming top-8.
// ---------------------------------------------------------------------------
__global__ __launch_bounds__(128) void knn_kernel(const _Float16* __restrict__ pts_h,
                                                  const _Float16* __restrict__ cent_h,
                                                  const float* __restrict__ norms,
                                                  const int* __restrict__ fps_ws,
                                                  int* __restrict__ knn_out) {
  const int b = blockIdx.x >> 6;
  const int srow0 = (blockIdx.x & 63) * 16;
  const int t = threadIdx.x;
  const int lane = t & 31, wave = t >> 5;

  __shared__ _Float16 sc[16 * DIM];      // centroid tile (f16), TDM-staged
  __shared__ float scn[16];              // centroid norms
  __shared__ float sdist[16 * 64];       // distance tile per chunk
  __shared__ float scd[16 * 64];         // candidate distances
  __shared__ int   sci[16 * 64];         // candidate indices

  if (t < 32) {  // wave 0 issues the TDM tile load and waits on TENSORcnt
    tdm_load_2d_f16((unsigned)(uintptr_t)sc,
                    cent_h + ((size_t)b * NSAMP + srow0) * DIM,
                    /*rows=*/16, /*rowlen=*/DIM, /*stride0=*/DIM);
    __builtin_amdgcn_s_wait_tensorcnt(0);
  }
  if (t < 16)
    scn[t] = norms[(size_t)b * NPTS + fps_ws[(size_t)b * NSAMP + srow0 + t]];
  __syncthreads();

  v16h a0 = load_a_f16(sc, DIM, 0, 0, lane);
  v16h a1 = load_a_f16(sc, DIM, 0, 32, lane);

  float bd[8]; int bi[8];
#pragma unroll
  for (int e = 0; e < 8; ++e) { bd[e] = 3.402823466e38f; bi[e] = 0; }

  const _Float16* pbase = pts_h + (size_t)b * NPTS * DIM;
  const float* nrm = norms + (size_t)b * NPTS;
  const int hi = lane >> 4, j = lane & 15;

  for (int n0 = 0; n0 < NPTS; n0 += 64) {
    if (n0 + 64 < NPTS)
      __builtin_prefetch(pbase + (size_t)(n0 + 64 + (t & 63)) * DIM, 0, 3);

    int nb = n0 + wave * 16;
    v16h b0 = load_b_nmajor(pbase, DIM, nb, 0, lane);
    v16h b1 = load_b_nmajor(pbase, DIM, nb, 32, lane);
    v8f c = {};
    c = wmma_f16(a0, b0, c);
    c = wmma_f16(a1, b1, c);

#pragma unroll
    for (int r = 0; r < 8; ++r) {
      int row = r + 8 * hi;          // centroid row 0..15
      int n = nb + j;                // point index within batch
      float dist = scn[row] + nrm[n] - 2.0f * c[r];
      sdist[row * 64 + wave * 16 + j] = dist;
    }
    __syncthreads();

    {
      int row = t >> 3, slot = t & 7;  // 8 merge threads per row
#pragma unroll
      for (int q = 0; q < 8; ++q) {
        int nl = slot * 8 + q;
        topk_insert(sdist[row * 64 + nl], n0 + nl, bd, bi);
      }
    }
    __syncthreads();
  }

  {
    int row = t >> 3, slot = t & 7;
#pragma unroll
    for (int e = 0; e < 8; ++e) {
      scd[row * 64 + slot * 8 + e] = bd[e];
      sci[row * 64 + slot * 8 + e] = bi[e];
    }
  }
  __syncthreads();

  if (t < 16) {
    float fd[8]; int fi[8];
#pragma unroll
    for (int e = 0; e < 8; ++e) { fd[e] = 3.402823466e38f; fi[e] = 0; }
    for (int q = 0; q < 64; ++q)
      topk_insert(scd[t * 64 + q], sci[t * 64 + q], fd, fi);
    int* dst = knn_out + ((size_t)b * NSAMP + srow0 + t) * KNN_K;
#pragma unroll
    for (int e = 0; e < 8; ++e) dst[e] = fi[e];
  }
}

// ---------------------------------------------------------------------------
// Kernel 6: fused pointwise MLP (64 -> 256 -> 256 -> 64, ReLU between).
// Block = 256 (8 waves) handles 32 points; X tile staged by the Tensor Data
// Mover; activations in LDS (f16); all GEMMs via WMMA 16x16x32 f16 + f32 acc.
// ---------------------------------------------------------------------------
__global__ __launch_bounds__(256) void mlp_kernel(const _Float16* __restrict__ xh,
                                                  const _Float16* __restrict__ w1h,
                                                  const float* __restrict__ b1,
                                                  const _Float16* __restrict__ w2h,
                                                  const float* __restrict__ b2,
                                                  const _Float16* __restrict__ w3h,
                                                  const float* __restrict__ b3,
                                                  float* __restrict__ out) {
  const int rows0 = blockIdx.x * 32;
  const int t = threadIdx.x;
  const int lane = t & 31, wave = t >> 5;
  const int hi = lane >> 4, nl = lane & 15;

  __shared__ _Float16 Xs[32 * DIM];  // input tile (f16), TDM-staged
  __shared__ _Float16 Hh[32 * HID];  // layer-1 activations (f16)
  __shared__ _Float16 Gh[32 * HID];  // layer-2 activations (f16)

  if (t < 32) {  // wave 0 issues the TDM tile load and waits on TENSORcnt
    tdm_load_2d_f16((unsigned)(uintptr_t)Xs, xh + (size_t)rows0 * DIM,
                    /*rows=*/32, /*rowlen=*/DIM, /*stride0=*/DIM);
    __builtin_amdgcn_s_wait_tensorcnt(0);
  }
  __syncthreads();

  // GEMM1: X(32x64) @ W1(64x256) + b1, ReLU -> Hh
#pragma unroll
  for (int tt = wave; tt < 32; tt += 8) {
    int slab = tt >> 4, col = tt & 15;
    v8f c = {};
#pragma unroll
    for (int ks = 0; ks < DIM; ks += 32) {
      v16h a  = load_a_f16(Xs, DIM, slab * 16, ks, lane);
      v16h bb = load_b_kmajor(w1h, HID, col * 16, ks, lane);
      c = wmma_f16(a, bb, c);
    }
#pragma unroll
    for (int r = 0; r < 8; ++r) {
      int row = slab * 16 + r + 8 * hi;
      int n = col * 16 + nl;
      float v = c[r] + b1[n];
      v = v > 0.f ? v : 0.f;
      Hh[row * HID + n] = (_Float16)v;
    }
  }
  __syncthreads();

  // GEMM2: Hh(32x256) @ W2(256x256) + b2, ReLU -> Gh
#pragma unroll
  for (int tt = wave; tt < 32; tt += 8) {
    int slab = tt >> 4, col = tt & 15;
    v8f c = {};
#pragma unroll
    for (int ks = 0; ks < HID; ks += 32) {
      v16h a  = load_a_f16(Hh, HID, slab * 16, ks, lane);
      v16h bb = load_b_kmajor(w2h, HID, col * 16, ks, lane);
      c = wmma_f16(a, bb, c);
    }
#pragma unroll
    for (int r = 0; r < 8; ++r) {
      int row = slab * 16 + r + 8 * hi;
      int n = col * 16 + nl;
      float v = c[r] + b2[n];
      v = v > 0.f ? v : 0.f;
      Gh[row * HID + n] = (_Float16)v;
    }
  }
  __syncthreads();

  // GEMM3: Gh(32x256) @ W3(256x64) + b3 -> out (one 16x16 tile per wave)
  {
    int slab = wave >> 2, col = wave & 3;
    v8f c = {};
#pragma unroll
    for (int ks = 0; ks < HID; ks += 32) {
      v16h a  = load_a_f16(Gh, HID, slab * 16, ks, lane);
      v16h bb = load_b_kmajor(w3h, DIM, col * 16, ks, lane);
      c = wmma_f16(a, bb, c);
    }
#pragma unroll
    for (int r = 0; r < 8; ++r) {
      int row = rows0 + slab * 16 + r + 8 * hi;
      int n = col * 16 + nl;
      out[(size_t)row * DIM + n] = c[r] + b3[n];
    }
  }
}

// ---------------------------------------------------------------------------
// Launch
// ---------------------------------------------------------------------------
extern "C" void kernel_launch(void* const* d_in, const int* in_sizes, int n_in,
                              void* d_out, int out_size, void* d_ws, size_t ws_size,
                              hipStream_t stream) {
  (void)in_sizes; (void)n_in; (void)out_size; (void)ws_size;

  const float* pts = (const float*)d_in[0];
  const float* W1  = (const float*)d_in[1];
  const float* b1  = (const float*)d_in[2];
  const float* W2  = (const float*)d_in[3];
  const float* b2  = (const float*)d_in[4];
  const float* W3  = (const float*)d_in[5];
  const float* b3  = (const float*)d_in[6];

  // Workspace layout (bytes, 256-aligned); total ~10.6 MB
  char* ws = (char*)d_ws;
  int*      fps_ws = (int*)(ws + 0);                  //  64 KB
  float*    norms  = (float*)(ws + 65536);            // 256 KB
  _Float16* pts_h  = (_Float16*)(ws + 327680);        //   8 MB
  _Float16* cent_h = (_Float16*)(ws + 8716288);       //   2 MB
  _Float16* w1h    = (_Float16*)(ws + 10813440);      //  32 KB
  _Float16* w2h    = (_Float16*)(ws + 10846208);      // 128 KB
  _Float16* w3h    = (_Float16*)(ws + 10977280);      //  32 KB

  float* out     = (float*)d_out;
  int*   outi    = (int*)d_out;
  int*   knn_out = outi + OUT_ELEMS;
  int*   fps_out = outi + OUT_ELEMS + KNN_ELEMS;

  prep_points <<<(BATCH * NPTS) / 256, 256, 0, stream>>>(pts, pts_h, norms);
  prep_weights<<<384, 256, 0, stream>>>(W1, W2, W3, w1h, w2h, w3h);
  fps_kernel  <<<BATCH, 1024, 0, stream>>>(pts, fps_ws, fps_out);
  gather_cent <<<(BATCH * NSAMP) / 256, 256, 0, stream>>>(fps_ws, pts_h, cent_h);
  knn_kernel  <<<BATCH * (NSAMP / 16), 128, 0, stream>>>(pts_h, cent_h, norms,
                                                         fps_ws, knn_out);
  mlp_kernel  <<<(BATCH * NPTS) / 32, 256, 0, stream>>>(pts_h, w1h, b1, w2h, b2,
                                                        w3h, b3, out);
}
